// NestedTensorBlock_27255862461004
// MI455X (gfx1250) — matrix operand order repaired and checked by reading, not compiled, not run on hardware
//
#include <hip/hip_runtime.h>
#include <hip/hip_bf16.h>

#define DMODEL 1024
#define NHEAD 16
#define DHEAD 64
#define ROWS_TOTAL 12288   // 8*1024 + 16*256
#define ROWS_X1 8192
#define EPSV 1e-5f

typedef __attribute__((ext_vector_type(16))) __bf16 v16bf;
typedef __attribute__((ext_vector_type(8)))  float  v8f;
typedef __attribute__((ext_vector_type(4)))  int    v4i;

union FragBF {
  v16bf v;
  unsigned int u[8];
  unsigned short s[16];
};

#if defined(__AMDGCN__) && __has_builtin(__builtin_amdgcn_global_load_async_to_lds_b128)
#define HAVE_ASYNC 1
#else
#define HAVE_ASYNC 0
#endif

// async copy builtin takes (v4i as1* src, v4i as3* dst, imm offset, imm cpol)
#define GPTR(p) ((__attribute__((address_space(1))) v4i*)(p))
#define LPTR(p) ((__attribute__((address_space(3))) v4i*)(p))

#if defined(__AMDGCN__) && __has_builtin(__builtin_amdgcn_s_wait_asynccnt)
#define WAIT_ASYNC() __builtin_amdgcn_s_wait_asynccnt(0)
#elif defined(__AMDGCN__)
#define WAIT_ASYNC() asm volatile("s_wait_asynccnt 0x0" ::: "memory")
#else
#define WAIT_ASYNC()
#endif

__device__ __forceinline__ unsigned short f2bf(float f) {
  unsigned int x = __float_as_uint(f);
  x = x + 0x7FFFu + ((x >> 16) & 1u);   // round-to-nearest-even
  return (unsigned short)(x >> 16);
}

// A-matrix 16x32 bf16 layout: pair p start-k for lane-half kh (contiguous k = 16*kh + 2p, matching B)
__device__ __forceinline__ int a_pair_k(int p, int kh) {
  return (p << 1) + (kh << 4);
}

__device__ __forceinline__ v8f vzero8() {
  v8f z = {0.f, 0.f, 0.f, 0.f, 0.f, 0.f, 0.f, 0.f};
  return z;
}

// ---------------- concat x1,x2 -> xcat (f32 residual stream) ----------------
__global__ void concat_kernel(const float* __restrict__ x1, const float* __restrict__ x2,
                              float* __restrict__ xcat, int n1, int ntot) {
  int i = blockIdx.x * blockDim.x + threadIdx.x;  // float4 index
  int e = i * 4;
  if (e >= ntot) return;
  float4 val;
  if (e < n1) val = ((const float4*)x1)[i];
  else        val = ((const float4*)x2)[(e - n1) >> 2];
  ((float4*)xcat)[i] = val;
}

// ---------------- LayerNorm(row of 1024) -> bf16 ----------------
__global__ void ln_bf16_kernel(const float* __restrict__ x, const float* __restrict__ g,
                               const float* __restrict__ be, unsigned short* __restrict__ out) {
  long row = blockIdx.x;
  long base = row * DMODEL;
  int i0 = threadIdx.x * 4;
  float v[4];
  float s = 0.f, ss = 0.f;
#pragma unroll
  for (int j = 0; j < 4; ++j) { v[j] = x[base + i0 + j]; s += v[j]; ss += v[j] * v[j]; }
#pragma unroll
  for (int m = 16; m >= 1; m >>= 1) { s += __shfl_xor(s, m); ss += __shfl_xor(ss, m); }
  __shared__ float sb[8], ssb[8];
  int w = threadIdx.x >> 5;
  if ((threadIdx.x & 31) == 0) { sb[w] = s; ssb[w] = ss; }
  __syncthreads();
  float ts = 0.f, tss = 0.f;
#pragma unroll
  for (int k = 0; k < 8; ++k) { ts += sb[k]; tss += ssb[k]; }
  float mu = ts * (1.f / DMODEL);
  float var = tss * (1.f / DMODEL) - mu * mu;
  float rstd = rsqrtf(var + EPSV);
#pragma unroll
  for (int j = 0; j < 4; ++j) {
    int idx = i0 + j;
    out[base + idx] = f2bf((v[j] - mu) * rstd * g[idx] + be[idx]);
  }
}

// ---------------- weight convert + transpose: wt[n*K+k] = bf16(w[k*N+n]) ----------------
__global__ void wt_bf16_kernel(const float* __restrict__ w, unsigned short* __restrict__ wt,
                               int K, int N) {
  int n = blockIdx.x * 32 + threadIdx.x;
  int k = blockIdx.y * 8 + threadIdx.y;
  if (n < N && k < K) wt[(long)n * K + k] = f2bf(w[(long)k * N + n]);
}

// ---------------- WMMA GEMM with async-to-LDS double-buffered pipeline ----------------
// C[M,N] = A[M,K](bf16) * W^T[N,K](bf16) + bias (+res)(+gelu)
// Block: 256 threads = 8 waves (2x4), tile 64(M) x 128(N), K-step 64.
// Each wave computes 32x32 via 2 A-frags x 2 B-frags = 4 WMMAs per 32-K substep.
template <bool RES, bool GELU, bool OUTBF16>
__global__ __launch_bounds__(256) void gemm_wmma_kernel(
    const unsigned short* __restrict__ A,
    const unsigned short* __restrict__ Bt,
    const float* __restrict__ bias,
    const float* __restrict__ res,
    void* __restrict__ outp,
    int K, int N) {
  __shared__ unsigned short As[2][64 * 64];    // [buf][row][k]  8KB each
  __shared__ unsigned short Bs[2][128 * 64];   // [buf][n][k]   16KB each

  int tid = threadIdx.x;
  int lane = tid & 31, wave = tid >> 5;
  int wm = wave >> 2, wn = wave & 3;           // 2 x 4 waves
  int kh = lane >> 4, ln = lane & 15;
  long tileM = (long)blockIdx.y * 64;
  long tileN = (long)blockIdx.x * 128;

  v8f acc00 = vzero8(), acc01 = vzero8(), acc10 = vzero8(), acc11 = vzero8();
  const int nK = K >> 6;

  // staging geometry: A = 512 16B-segments (2/thread), B = 1024 (4/thread)
  int rowA[2], skA[2], rowB[4], skB[4];
#pragma unroll
  for (int i = 0; i < 2; ++i) { int s = tid + (i << 8); rowA[i] = s >> 3; skA[i] = (s & 7) << 3; }
#pragma unroll
  for (int i = 0; i < 4; ++i) { int s = tid + (i << 8); rowB[i] = s >> 3; skB[i] = (s & 7) << 3; }

#if HAVE_ASYNC
  // -------- prologue: stage tile 0 --------
#pragma unroll
  for (int i = 0; i < 2; ++i)
    __builtin_amdgcn_global_load_async_to_lds_b128(
        GPTR(A + (tileM + rowA[i]) * (long)K + skA[i]),
        LPTR(&As[0][rowA[i] * 64 + skA[i]]), 0, 0);
#pragma unroll
  for (int i = 0; i < 4; ++i)
    __builtin_amdgcn_global_load_async_to_lds_b128(
        GPTR(Bt + (tileN + rowB[i]) * (long)K + skB[i]),
        LPTR(&Bs[0][rowB[i] * 64 + skB[i]]), 0, 0);
  WAIT_ASYNC();
  __syncthreads();

  for (int kt = 0; kt < nK; ++kt) {
    int cur = kt & 1;
    if (kt + 1 < nK) {                     // stage next tile into other buffer
      int k0 = (kt + 1) << 6;
#pragma unroll
      for (int i = 0; i < 2; ++i)
        __builtin_amdgcn_global_load_async_to_lds_b128(
            GPTR(A + (tileM + rowA[i]) * (long)K + k0 + skA[i]),
            LPTR(&As[cur ^ 1][rowA[i] * 64 + skA[i]]), 0, 0);
#pragma unroll
      for (int i = 0; i < 4; ++i)
        __builtin_amdgcn_global_load_async_to_lds_b128(
            GPTR(Bt + (tileN + rowB[i]) * (long)K + k0 + skB[i]),
            LPTR(&Bs[cur ^ 1][rowB[i] * 64 + skB[i]]), 0, 0);
    }
    // -------- compute on current buffer --------
#pragma unroll
    for (int ks = 0; ks < 64; ks += 32) {
      FragBF a0, a1, b0, b1;
#pragma unroll
      for (int p = 0; p < 8; ++p) {
        int kk = ks + a_pair_k(p, kh);
        a0.u[p] = *(const unsigned int*)&As[cur][(wm * 32 + ln) * 64 + kk];
        a1.u[p] = *(const unsigned int*)&As[cur][(wm * 32 + 16 + ln) * 64 + kk];
      }
#pragma unroll
      for (int p = 0; p < 8; ++p) {
        int kk = ks + (kh << 4) + (p << 1);
        b0.u[p] = *(const unsigned int*)&Bs[cur][(wn * 32 + ln) * 64 + kk];
        b1.u[p] = *(const unsigned int*)&Bs[cur][(wn * 32 + 16 + ln) * 64 + kk];
      }
      acc00 = __builtin_amdgcn_wmma_f32_16x16x32_bf16(false, a0.v, false, b0.v, (short)0, acc00, false, false);
      acc01 = __builtin_amdgcn_wmma_f32_16x16x32_bf16(false, a0.v, false, b1.v, (short)0, acc01, false, false);
      acc10 = __builtin_amdgcn_wmma_f32_16x16x32_bf16(false, a1.v, false, b0.v, (short)0, acc10, false, false);
      acc11 = __builtin_amdgcn_wmma_f32_16x16x32_bf16(false, a1.v, false, b1.v, (short)0, acc11, false, false);
    }
    WAIT_ASYNC();
    __syncthreads();
  }
#else
  // -------- synchronous fallback: regs -> LDS with two barriers --------
  for (int kt = 0; kt < nK; ++kt) {
    int k0 = kt << 6;
    uint4 ra[2], rb[4];
#pragma unroll
    for (int i = 0; i < 2; ++i)
      ra[i] = *(const uint4*)&A[(tileM + rowA[i]) * (long)K + k0 + skA[i]];
#pragma unroll
    for (int i = 0; i < 4; ++i)
      rb[i] = *(const uint4*)&Bt[(tileN + rowB[i]) * (long)K + k0 + skB[i]];
    __syncthreads();
#pragma unroll
    for (int i = 0; i < 2; ++i) *(uint4*)&As[0][rowA[i] * 64 + skA[i]] = ra[i];
#pragma unroll
    for (int i = 0; i < 4; ++i) *(uint4*)&Bs[0][rowB[i] * 64 + skB[i]] = rb[i];
    __syncthreads();
#pragma unroll
    for (int ks = 0; ks < 64; ks += 32) {
      FragBF a0, a1, b0, b1;
#pragma unroll
      for (int p = 0; p < 8; ++p) {
        int kk = ks + a_pair_k(p, kh);
        a0.u[p] = *(const unsigned int*)&As[0][(wm * 32 + ln) * 64 + kk];
        a1.u[p] = *(const unsigned int*)&As[0][(wm * 32 + 16 + ln) * 64 + kk];
      }
#pragma unroll
      for (int p = 0; p < 8; ++p) {
        int kk = ks + (kh << 4) + (p << 1);
        b0.u[p] = *(const unsigned int*)&Bs[0][(wn * 32 + ln) * 64 + kk];
        b1.u[p] = *(const unsigned int*)&Bs[0][(wn * 32 + 16 + ln) * 64 + kk];
      }
      acc00 = __builtin_amdgcn_wmma_f32_16x16x32_bf16(false, a0.v, false, b0.v, (short)0, acc00, false, false);
      acc01 = __builtin_amdgcn_wmma_f32_16x16x32_bf16(false, a0.v, false, b1.v, (short)0, acc01, false, false);
      acc10 = __builtin_amdgcn_wmma_f32_16x16x32_bf16(false, a1.v, false, b0.v, (short)0, acc10, false, false);
      acc11 = __builtin_amdgcn_wmma_f32_16x16x32_bf16(false, a1.v, false, b1.v, (short)0, acc11, false, false);
    }
  }
#endif

  // -------- epilogue --------
  long n0 = tileN + wn * 32 + ln;
  float bb0 = bias[n0];
  float bb1 = bias[n0 + 16];
#pragma unroll
  for (int r = 0; r < 8; ++r) {
    long row0 = tileM + wm * 32 + r + (kh << 3);
    long row1 = row0 + 16;
    float v00 = acc00[r] + bb0, v01 = acc01[r] + bb1;
    float v10 = acc10[r] + bb0, v11 = acc11[r] + bb1;
    if (GELU) {
      v00 = v00 / (1.f + __expf(-1.702f * v00));
      v01 = v01 / (1.f + __expf(-1.702f * v01));
      v10 = v10 / (1.f + __expf(-1.702f * v10));
      v11 = v11 / (1.f + __expf(-1.702f * v11));
    }
    if (RES) {
      v00 += res[row0 * N + n0];
      v01 += res[row0 * N + n0 + 16];
      v10 += res[row1 * N + n0];
      v11 += res[row1 * N + n0 + 16];
    }
    if (OUTBF16) {
      unsigned short* o = (unsigned short*)outp;
      o[row0 * N + n0]      = f2bf(v00);
      o[row0 * N + n0 + 16] = f2bf(v01);
      o[row1 * N + n0]      = f2bf(v10);
      o[row1 * N + n0 + 16] = f2bf(v11);
    } else {
      float* o = (float*)outp;
      o[row0 * N + n0]      = v00;
      o[row0 * N + n0 + 16] = v01;
      o[row1 * N + n0]      = v10;
      o[row1 * N + n0 + 16] = v11;
    }
  }
}

// ---------------- flash attention: 1 wave per 16-query tile per (seq, head) ----------------
// __launch_bounds__(32, 1): single wave, min occupancy 1 -> full VGPR budget, no spills.
__global__ __launch_bounds__(32, 1) void attn_wmma_kernel(
    const unsigned short* __restrict__ qkv,
    unsigned short* __restrict__ out,
    int row0, int N) {
  const int LD = 3 * DMODEL;
  const long STEP = 32L * LD;              // pointer advance per key tile (elements)
  int lane = threadIdx.x;
  int kh = lane >> 4, ln = lane & 15;
  int h = blockIdx.y;
  long seqBase = row0 + (long)blockIdx.z * N;
  long qRow = seqBase + blockIdx.x * 16;
  int qc = h * DHEAD;
  int kc = DMODEL + h * DHEAD;
  int vc = 2 * DMODEL + h * DHEAD;

  // Q fragments for K-dim 0..31 and 32..63 (held for the whole key loop)
  FragBF aq0, aq1;
  {
    const unsigned short* qr = qkv + (qRow + ln) * (long)LD + qc;
#pragma unroll
    for (int p = 0; p < 8; ++p) {
      aq0.u[p] = *(const unsigned int*)&qr[a_pair_k(p, kh)];
      aq1.u[p] = *(const unsigned int*)&qr[32 + a_pair_k(p, kh)];
    }
  }

  // Hoisted per-lane global pointers, advanced by STEP each key tile.
  const unsigned short* kr0 = qkv + (seqBase + ln) * (long)LD + kc;
  const unsigned short* kr1 = kr0 + 16L * LD;
#if HAVE_ASYNC
  // segment c of the V tile: row (lane>>3)+4c, byte-col (lane&7)*16
  // -> one base pointer, constant element offsets c*4*LD (global) / c*256 (LDS)
  const unsigned short* vp0 = qkv + (seqBase + (lane >> 3)) * (long)LD + vc + ((lane & 7) << 3);
  const int vlds0 = (lane >> 3) * 64 + ((lane & 7) << 3);
#endif

  float mrow[8], lrow[8];
#pragma unroll
  for (int r = 0; r < 8; ++r) { mrow[r] = -3.0e38f; lrow[r] = 0.f; }
  v8f o0 = vzero8(), o1 = vzero8(), o2 = vzero8(), o3 = vzero8();

  __shared__ unsigned short P[16 * 32];    // P tile staging (C-layout -> A-layout)
  __shared__ unsigned short Vs[32 * 64];   // V tile staging [key_local][col 0..63]

  for (int kt = 0; kt < N; kt += 32) {
#if HAVE_ASYNC
    // stage V tile (32 keys x 64 cols) via async copies; overlaps with S compute
#pragma unroll
    for (int c = 0; c < 8; ++c)
      __builtin_amdgcn_global_load_async_to_lds_b128(
          GPTR(vp0 + (long)c * 4 * LD), LPTR(&Vs[vlds0 + c * 256]), 0, 0);
    vp0 += STEP;
#endif
    v8f s0 = vzero8(), s1 = vzero8();
    FragBF bk;
    // S0 = Q * K^T (keys kt..kt+15), two K-dim steps of 32
#pragma unroll
    for (int p = 0; p < 8; ++p) bk.u[p] = *(const unsigned int*)&kr0[(kh << 4) + (p << 1)];
    s0 = __builtin_amdgcn_wmma_f32_16x16x32_bf16(false, aq0.v, false, bk.v, (short)0, s0, false, false);
#pragma unroll
    for (int p = 0; p < 8; ++p) bk.u[p] = *(const unsigned int*)&kr0[32 + (kh << 4) + (p << 1)];
    s0 = __builtin_amdgcn_wmma_f32_16x16x32_bf16(false, aq1.v, false, bk.v, (short)0, s0, false, false);
    // S1 = keys kt+16..kt+31
#pragma unroll
    for (int p = 0; p < 8; ++p) bk.u[p] = *(const unsigned int*)&kr1[(kh << 4) + (p << 1)];
    s1 = __builtin_amdgcn_wmma_f32_16x16x32_bf16(false, aq0.v, false, bk.v, (short)0, s1, false, false);
#pragma unroll
    for (int p = 0; p < 8; ++p) bk.u[p] = *(const unsigned int*)&kr1[32 + (kh << 4) + (p << 1)];
    s1 = __builtin_amdgcn_wmma_f32_16x16x32_bf16(false, aq1.v, false, bk.v, (short)0, s1, false, false);
    kr0 += STEP;
    kr1 += STEP;

    // online softmax over the 32 new keys; row r lives on 16 lanes of this half-group
#pragma unroll
    for (int r = 0; r < 8; ++r) {
      float t = fmaxf(s0[r], s1[r]) * 0.125f;            // 1/sqrt(64)
#pragma unroll
      for (int m = 8; m >= 1; m >>= 1) t = fmaxf(t, __shfl_xor(t, m));
      float mn = fmaxf(mrow[r], t);
      float alpha = __expf(mrow[r] - mn);
      float p0 = __expf(s0[r] * 0.125f - mn);
      float p1 = __expf(s1[r] * 0.125f - mn);
      float rs = p0 + p1;
#pragma unroll
      for (int m = 8; m >= 1; m >>= 1) rs += __shfl_xor(rs, m);
      lrow[r] = lrow[r] * alpha + rs;
      mrow[r] = mn;
      o0[r] *= alpha; o1[r] *= alpha; o2[r] *= alpha; o3[r] *= alpha;
      int prow = r + (kh << 3);
      P[prow * 32 + ln] = f2bf(p0);
      P[prow * 32 + 16 + ln] = f2bf(p1);
    }
    // reload P in A-fragment layout (single wave: ds waits inserted by compiler)
    FragBF ap;
#pragma unroll
    for (int p = 0; p < 8; ++p)
      ap.u[p] = *(const unsigned int*)&P[ln * 32 + a_pair_k(p, kh)];
#if HAVE_ASYNC
    WAIT_ASYNC();   // V tile landed in LDS
#endif
    // O += P * V   (4 column tiles of 16 over DHEAD=64)
#pragma unroll
    for (int c = 0; c < 4; ++c) {
      FragBF bvv;
#pragma unroll
      for (int j = 0; j < 16; ++j) {
#if HAVE_ASYNC
        bvv.s[j] = Vs[((kh << 4) + j) * 64 + c * 16 + ln];
#else
        bvv.s[j] = qkv[(seqBase + kt + (kh << 4) + j) * (long)LD + vc + c * 16 + ln];
#endif
      }
      v8f& oc = (c == 0 ? o0 : c == 1 ? o1 : c == 2 ? o2 : o3);
      oc = __builtin_amdgcn_wmma_f32_16x16x32_bf16(false, ap.v, false, bvv.v,
                                                   (short)0, oc, false, false);
    }
  }

#pragma unroll
  for (int r = 0; r < 8; ++r) {
    float inv = 1.f / lrow[r];
    long row = qRow + r + (kh << 3);
    unsigned short* orow = out + row * DMODEL + h * DHEAD;
    orow[0 + ln]  = f2bf(o0[r] * inv);
    orow[16 + ln] = f2bf(o1[r] * inv);
    orow[32 + ln] = f2bf(o2[r] * inv);
    orow[48 + ln] = f2bf(o3[r] * inv);
  }
}

extern "C" void kernel_launch(void* const* d_in, const int* in_sizes, int n_in,
                              void* d_out, int out_size, void* d_ws, size_t ws_size,
                              hipStream_t stream) {
  (void)in_sizes; (void)n_in; (void)out_size; (void)ws_size;
  const float* x1    = (const float*)d_in[0];
  const float* x2    = (const float*)d_in[1];
  const float* w_qkv = (const float*)d_in[2];
  const float* b_qkv = (const float*)d_in[3];
  const float* w_o   = (const float*)d_in[4];
  const float* b_o   = (const float*)d_in[5];
  const float* g1    = (const float*)d_in[6];
  const float* be1   = (const float*)d_in[7];
  const float* w_fc  = (const float*)d_in[8];
  const float* b_fc  = (const float*)d_in[9];
  const float* w_pr  = (const float*)d_in[10];
  const float* b_pr  = (const float*)d_in[11];
  const float* g2    = (const float*)d_in[12];
  const float* be2   = (const float*)d_in[13];

  char* p = (char*)d_ws;
  float* xcat = (float*)p;            p += (size_t)ROWS_TOTAL * 1024 * 4;
  float* xres = (float*)p;            p += (size_t)ROWS_TOTAL * 1024 * 4;
  unsigned short* xln  = (unsigned short*)p; p += (size_t)ROWS_TOTAL * 1024 * 2;
  unsigned short* qkvb = (unsigned short*)p; p += (size_t)ROWS_TOTAL * 3072 * 2;
  unsigned short* attnb= (unsigned short*)p; p += (size_t)ROWS_TOTAL * 1024 * 2;
  unsigned short* ub   = (unsigned short*)p; p += (size_t)ROWS_TOTAL * 4096 * 2;
  unsigned short* wtq  = (unsigned short*)p; p += (size_t)3072 * 1024 * 2;
  unsigned short* wto  = (unsigned short*)p; p += (size_t)1024 * 1024 * 2;
  unsigned short* wtf  = (unsigned short*)p; p += (size_t)4096 * 1024 * 2;
  unsigned short* wtp  = (unsigned short*)p; p += (size_t)1024 * 4096 * 2;

  dim3 cb(32, 8);
  wt_bf16_kernel<<<dim3(3072 / 32, 1024 / 8), cb, 0, stream>>>(w_qkv, wtq, 1024, 3072);
  wt_bf16_kernel<<<dim3(1024 / 32, 1024 / 8), cb, 0, stream>>>(w_o,   wto, 1024, 1024);
  wt_bf16_kernel<<<dim3(4096 / 32, 1024 / 8), cb, 0, stream>>>(w_fc,  wtf, 1024, 4096);
  wt_bf16_kernel<<<dim3(1024 / 32, 4096 / 8), cb, 0, stream>>>(w_pr,  wtp, 4096, 1024);

  int ntot = ROWS_TOTAL * 1024;
  concat_kernel<<<(ntot / 4 + 255) / 256, 256, 0, stream>>>(x1, x2, xcat, ROWS_X1 * 1024, ntot);

  // ---- LN1 + QKV ----
  ln_bf16_kernel<<<ROWS_TOTAL, 256, 0, stream>>>(xcat, g1, be1, xln);
  gemm_wmma_kernel<false, false, true>
      <<<dim3(3072 / 128, ROWS_TOTAL / 64), 256, 0, stream>>>(xln, wtq, b_qkv, nullptr, qkvb, 1024, 3072);

  // ---- attention (per-sequence, block-diagonal) ----
  attn_wmma_kernel<<<dim3(1024 / 16, NHEAD, 8), 32, 0, stream>>>(qkvb, attnb, 0, 1024);
  attn_wmma_kernel<<<dim3(256 / 16, NHEAD, 16), 32, 0, stream>>>(qkvb, attnb, ROWS_X1, 256);

  // ---- O projection + residual ----
  gemm_wmma_kernel<true, false, false>
      <<<dim3(1024 / 128, ROWS_TOTAL / 64), 256, 0, stream>>>(attnb, wto, b_o, xcat, xres, 1024, 1024);

  // ---- LN2 + MLP ----
  ln_bf16_kernel<<<ROWS_TOTAL, 256, 0, stream>>>(xres, g2, be2, xln);
  gemm_wmma_kernel<false, true, true>
      <<<dim3(4096 / 128, ROWS_TOTAL / 64), 256, 0, stream>>>(xln, wtf, b_fc, nullptr, ub, 1024, 4096);
  gemm_wmma_kernel<true, false, false>
      <<<dim3(1024 / 128, ROWS_TOTAL / 64), 256, 0, stream>>>(ub, wtp, b_pr, xres, (float*)d_out, 4096, 1024);
}
